// DeformableAlignment_17360257811041
// MI455X (gfx1250) — matrix operand order repaired and checked by reading, not compiled, not run on hardware
//
#include <hip/hip_runtime.h>
#include <hip/hip_bf16.h>
#include <math.h>

// Problem constants (from reference)
#define BB   8
#define HH   96
#define WW   160
#define HP   (HH + 2)     // spatially padded (1-px zero halo) -> branch-free taps
#define WP   (WW + 2)
#define NPIX (BB*HH*WW)   // 122880 pixels, multiple of 128
#define CIN0 196          // 64*3 + 4 concat channels
#define CP0  224          // padded to 7*32 for K-chunks of 32
#define OC4  216          // K*K*3*G valid channels of layer 4
#define NP4  256          // padded to 16 N-tiles of 16 (all blocks run 4 full tiles)

typedef __attribute__((ext_vector_type(16))) _Float16 v16h;
typedef __attribute__((ext_vector_type(8)))  float    v8f;

static __device__ __forceinline__ unsigned short f32_to_h(float x) {
  union { _Float16 f; unsigned short u; } c; c.f = (_Float16)x; return c.u;
}

// ---------------------------------------------------------------------------
// Kernel 0: fused concat -> halo-padded NHWC f16 buffer [B][HP][WP][224].
// Halo rows/cols and channels >=196 are written as zero.
// ---------------------------------------------------------------------------
__global__ void pack_cat_kernel(const float* __restrict__ x0, const float* __restrict__ x1,
                                const float* __restrict__ x2, const float* __restrict__ f1,
                                const float* __restrict__ f2, _Float16* __restrict__ out) {
  long idx = (long)blockIdx.x * 256 + threadIdx.x;
  if (idx >= (long)BB * HP * WP * CP0) return;
  int c    = (int)(idx % CP0);
  long pp  = idx / CP0;
  int b    = (int)(pp / (HP * WP));
  int r    = (int)(pp % (HP * WP));
  int hy   = r / WP - 1;
  int wx   = r % WP - 1;
  float v = 0.f;
  if (hy >= 0 && hy < HH && wx >= 0 && wx < WW) {
    const int hw = hy * WW + wx;
    if      (c < 64)  v = x0[(size_t)(b * 64 + c      ) * HH * WW + hw];
    else if (c < 128) v = x1[(size_t)(b * 64 + c - 64 ) * HH * WW + hw];
    else if (c < 192) v = x2[(size_t)(b * 64 + c - 128) * HH * WW + hw];
    else if (c < 194) v = f1[(size_t)(b * 2  + c - 192) * HH * WW + hw];
    else if (c < 196) v = f2[(size_t)(b * 2  + c - 194) * HH * WW + hw];
  }
  out[idx] = (_Float16)v;
}

// Zero an f16 buffer (for act1-3 halos; d_ws is poisoned, halos must be 0).
__global__ void zero_f16_kernel(unsigned int* __restrict__ p, long ndw) {
  long i = (long)blockIdx.x * 256 + threadIdx.x;
  if (i < ndw) p[i] = 0u;
}

// ---------------------------------------------------------------------------
// Kernel 1: repack OIHW f32 weights into per-lane WMMA B-fragment layout:
//   wpk[tap][kc][nt][lane(32)][dword(8)]
// B element (k,n) of the 32x16 f16 B matrix: lane = (n%16)+16*((k/8)%2),
// half index = (k%8)+8*(k/16) -> dword v holds halves (2v, 2v+1).
// ---------------------------------------------------------------------------
__global__ void pack_w_kernel(const float* __restrict__ w, unsigned int* __restrict__ wpk,
                              int Cin, int KC, int Nout, int NT) {
  int d = blockIdx.x * 256 + threadIdx.x;
  int total = 9 * KC * NT * 256;
  if (d >= total) return;
  int v    = d & 7;
  int lane = (d >> 3) & 31;
  int blkI = d >> 8;                 // (tap*KC + kc)*NT + nt
  int nt   = blkI % NT;
  int kc   = (blkI / NT) % KC;
  int tap  = blkI / (NT * KC);
  int n    = (lane & 15) + nt * 16;
  int k0   = 2 * v + 8 * (lane >> 4) + 8 * (v >> 2);
  int c0   = kc * 32 + k0;
  unsigned short h0 = 0, h1 = 0;
  if (n < Nout && c0     < Cin) h0 = f32_to_h(w[((size_t)n * Cin + c0    ) * 9 + tap]);
  if (n < Nout && c0 + 1 < Cin) h1 = f32_to_h(w[((size_t)n * Cin + c0 + 1) * 9 + tap]);
  wpk[d] = (unsigned)h0 | ((unsigned)h1 << 16);
}

// ---------------------------------------------------------------------------
// Kernel 2: implicit-GEMM 3x3 conv via V_WMMA_F32_16X16X32_F16.
// Block = 128 threads = 4 waves. Each wave owns 32 consecutive pixels
// (two M=16 tiles) x 64 output channels (4 N-tiles). Per K-step:
// 4 A b128 loads + 8 B b128 loads + 8 WMMAs, fully branch-free
// (spatial halo makes every tap address valid; EXEC stays all-ones).
//
// A fragment (16-bit A 16x32 layout) loads straight from halo-padded NHWC:
// lane L (pixel m=L%16) needs halves {0..7,16..23} (L<16) or {8..15,24..31}
// (L>=16): two aligned b128 chunks at half-offsets koff and koff+16,
// koff = 8*(L>=16).
// ---------------------------------------------------------------------------
template <int KC, bool LRELU, bool OUT32>
__global__ __launch_bounds__(128)
void conv_gemm_kernel(const _Float16* __restrict__ in, const unsigned int* __restrict__ wpk,
                      const float* __restrict__ bias, _Float16* __restrict__ out16,
                      float* __restrict__ out32, int NTtot, int Nvalid, int NPout) {
  constexpr int CP = KC * 32;
  const int tid  = threadIdx.x;
  const int wv   = tid >> 5;
  const int lane = tid & 31;
  const int pixBase = (blockIdx.x * 4 + wv) * 32;    // 32-aligned -> single (b,h) row
  const int b  = pixBase / (HH * WW);
  const int hw = pixBase % (HH * WW);
  const int h  = hw / WW;
  const int w0 = hw % WW;
  const int nt0 = blockIdx.y * 4;                    // always 4 full N-tiles

  const int m    = lane & 15;
  const int hiL  = lane >> 4;
  const int koff = hiL * 8;                          // half-offset of first A chunk

  v8f a00 = {0,0,0,0,0,0,0,0}, a01 = {0,0,0,0,0,0,0,0};
  v8f a02 = {0,0,0,0,0,0,0,0}, a03 = {0,0,0,0,0,0,0,0};
  v8f a10 = {0,0,0,0,0,0,0,0}, a11 = {0,0,0,0,0,0,0,0};
  v8f a12 = {0,0,0,0,0,0,0,0}, a13 = {0,0,0,0,0,0,0,0};

  for (int tap = 0; tap < 9; ++tap) {
    const int ky = tap / 3, kx = tap % 3;
    // padded coords: hyp = h+ky, wxp = w0+m+kx (always in [0,HP)x[0,WP))
    const _Float16* arow = in +
        ((size_t)(b * HP + h + ky) * WP + (w0 + m + kx)) * CP + koff;
    const unsigned int* wb = wpk + (((size_t)tap * KC * NTtot + nt0) << 8) + lane * 8;

    __builtin_prefetch((const void*)(arow + (size_t)WP * CP), 0, 0);  // next tap row

    #pragma unroll
    for (int kc = 0; kc < KC; ++kc) {
      union { uint4 q[2]; v16h h; } af0, af1;
      const uint4* ap0 = (const uint4*)(arow + kc * 32);
      const uint4* ap1 = (const uint4*)(arow + kc * 32 + 16 * CP);  // pixels m+16
      af0.q[0] = ap0[0]; af0.q[1] = ap0[2];
      af1.q[0] = ap1[0]; af1.q[1] = ap1[2];

      const unsigned int* wk = wb + ((size_t)kc * NTtot << 8);
      union { uint4 q[2]; v16h h; } bf0, bf1, bf2, bf3;
      bf0.q[0] = ((const uint4*)(wk      ))[0]; bf0.q[1] = ((const uint4*)(wk      ))[1];
      bf1.q[0] = ((const uint4*)(wk + 256))[0]; bf1.q[1] = ((const uint4*)(wk + 256))[1];
      bf2.q[0] = ((const uint4*)(wk + 512))[0]; bf2.q[1] = ((const uint4*)(wk + 512))[1];
      bf3.q[0] = ((const uint4*)(wk + 768))[0]; bf3.q[1] = ((const uint4*)(wk + 768))[1];

      a00 = __builtin_amdgcn_wmma_f32_16x16x32_f16(false, af0.h, false, bf0.h, (short)0, a00, false, false);
      a01 = __builtin_amdgcn_wmma_f32_16x16x32_f16(false, af0.h, false, bf1.h, (short)0, a01, false, false);
      a02 = __builtin_amdgcn_wmma_f32_16x16x32_f16(false, af0.h, false, bf2.h, (short)0, a02, false, false);
      a03 = __builtin_amdgcn_wmma_f32_16x16x32_f16(false, af0.h, false, bf3.h, (short)0, a03, false, false);
      a10 = __builtin_amdgcn_wmma_f32_16x16x32_f16(false, af1.h, false, bf0.h, (short)0, a10, false, false);
      a11 = __builtin_amdgcn_wmma_f32_16x16x32_f16(false, af1.h, false, bf1.h, (short)0, a11, false, false);
      a12 = __builtin_amdgcn_wmma_f32_16x16x32_f16(false, af1.h, false, bf2.h, (short)0, a12, false, false);
      a13 = __builtin_amdgcn_wmma_f32_16x16x32_f16(false, af1.h, false, bf3.h, (short)0, a13, false, false);
    }
  }

  // Epilogue. D layout: lanes 0-15 -> M=v, N=lane; lanes 16-31 -> M=v+8, N=lane-16.
  const size_t prow = (size_t)(b * HP + h + 1) * WP + (w0 + 1);  // padded out base
  auto store_tile = [&](const v8f& a, int j, int mt) {
    const int n = (lane & 15) + (nt0 + j) * 16;
    if (n < Nvalid) {
      const float bv = bias[n];
      #pragma unroll
      for (int v = 0; v < 8; ++v) {
        const int mm = mt * 16 + v + 8 * hiL;
        float val = a[v] + bv;
        if (LRELU) val = (val >= 0.f) ? val : 0.1f * val;
        if (OUT32) out32[(size_t)(pixBase + mm) * NPout + n] = val;       // unpadded
        else       out16[(prow + mm) * NPout + n] = (_Float16)val;        // halo-padded
      }
    }
  };
  store_tile(a00, 0, 0); store_tile(a01, 1, 0); store_tile(a02, 2, 0); store_tile(a03, 3, 0);
  store_tile(a10, 0, 1); store_tile(a11, 1, 1); store_tile(a12, 2, 1); store_tile(a13, 3, 1);
}

// ---------------------------------------------------------------------------
// Kernel 3: deformable conv. Block = 256 threads = 4 pixels x 64 threads.
// Phase 1: build sampled[g*8+c][tap] (576 f32) in LDS with offset fixup
//          (offset += reversed tiled flow), sigmoid mask, bilinear gather.
// Phase 2: each thread = one output channel, 576-MAC dot with OIHW weights.
// ---------------------------------------------------------------------------
__global__ __launch_bounds__(256)
void deform_kernel(const float* __restrict__ x, const float* __restrict__ o4,
                   const float* __restrict__ flow1, const float* __restrict__ flow2,
                   const float* __restrict__ wdef, const float* __restrict__ bias,
                   float* __restrict__ out) {
  __shared__ float samp[4][576];
  const int sp  = threadIdx.x >> 6;
  const int t   = threadIdx.x & 63;
  const int pix = blockIdx.x * 4 + sp;
  const int b   = pix / (HH * WW);
  const int hw  = pix % (HH * WW);
  const int h   = hw / WW;
  const int w   = hw % WW;

  for (int pair = t; pair < 72; pair += 64) {
    const int g = pair / 9, tap = pair % 9;
    const int ky = tap / 3, kx = tap % 3;
    const float* fl = (g < 4) ? flow1 : flow2;
    // tile(flow[:, ::-1]): even offset channel (dy) gets flow ch1, odd gets ch0
    const float fy = fl[((size_t)(b * 2 + 1) * HH + h) * WW + w];
    const float fx = fl[((size_t)(b * 2 + 0) * HH + h) * WW + w];
    const int obase = (g < 4) ? (g * 9 + tap) * 2 : 72 + ((g - 4) * 9 + tap) * 2;
    const size_t orow = (size_t)pix * NP4;
    const float dy = o4[orow + obase]     + fy;
    const float dx = o4[orow + obase + 1] + fx;
    const float mk = 1.f / (1.f + expf(-o4[orow + 144 + g * 9 + tap]));
    const float ys = dy + (float)(h - 1 + ky);
    const float xs = dx + (float)(w - 1 + kx);
    const float y0f = floorf(ys), x0f = floorf(xs);
    const float wy1 = ys - y0f, wx1 = xs - x0f;
    const float wy0 = 1.f - wy1, wx0 = 1.f - wx1;
    const int y0 = (int)y0f, x0 = (int)x0f;
    #pragma unroll
    for (int c = 0; c < 8; ++c) {
      const int ch = g * 8 + c;
      const float* img = x + (size_t)(b * 64 + ch) * HH * WW;
      float s = 0.f;
      #pragma unroll
      for (int q = 0; q < 4; ++q) {
        const int yi = y0 + (q >> 1), xi = x0 + (q & 1);
        const float wq = ((q >> 1) ? wy1 : wy0) * ((q & 1) ? wx1 : wx0);
        if (yi >= 0 && yi < HH && xi >= 0 && xi < WW)
          s += img[yi * WW + xi] * wq;
      }
      samp[sp][ch * 9 + tap] = s * mk;
    }
  }
  __syncthreads();

  float accv = bias[t];
  const float* wrow = wdef + (size_t)t * 576;  // OIHW (64,64,3,3): oc*576 + ch*9 + tap
  for (int i = 0; i < 576; ++i) accv += samp[sp][i] * wrow[i];
  out[((size_t)(b * 64 + t) * HH + h) * WW + w] = accv;
}

// ---------------------------------------------------------------------------
// Host launcher. Workspace ~235 MB: act0 57MB, act1-3 16.3MB each (halo-
// padded), o4 (f32, 256ch) 126MB, packed weights <1MB.
// ---------------------------------------------------------------------------
extern "C" void kernel_launch(void* const* d_in, const int* in_sizes, int n_in,
                              void* d_out, int out_size, void* d_ws, size_t ws_size,
                              hipStream_t stream) {
  (void)in_sizes; (void)n_in; (void)out_size; (void)ws_size;
  const float* x0 = (const float*)d_in[0];   // current_feature_warped
  const float* x1 = (const float*)d_in[1];   // feat_prop
  const float* x2 = (const float*)d_in[2];   // feat_prev
  const float* f1 = (const float*)d_in[3];   // flow1
  const float* f2 = (const float*)d_in[4];   // flow2
  const float* w1 = (const float*)d_in[5];
  const float* b1 = (const float*)d_in[6];
  const float* w2 = (const float*)d_in[7];
  const float* b2 = (const float*)d_in[8];
  const float* w3 = (const float*)d_in[9];
  const float* b3 = (const float*)d_in[10];
  const float* w4 = (const float*)d_in[11];
  const float* b4 = (const float*)d_in[12];
  const float* wd = (const float*)d_in[13];  // deform weight
  const float* bd = (const float*)d_in[14];  // deform bias
  float* out = (float*)d_out;

  char* ws = (char*)d_ws;
  size_t off = 0;
  auto take = [&](size_t bytes) -> char* {
    char* p = ws + off;
    off += (bytes + 255) & ~(size_t)255;
    return p;
  };
  const size_t actElems = (size_t)BB * HP * WP * 64;   // halo-padded 64-ch f16
  _Float16*     act0 = (_Float16*)    take((size_t)BB * HP * WP * CP0 * 2);
  _Float16*     act1 = (_Float16*)    take(actElems * 2);
  _Float16*     act2 = (_Float16*)    take(actElems * 2);
  _Float16*     act3 = (_Float16*)    take(actElems * 2);
  float*        o4   = (float*)       take((size_t)NPIX * NP4 * 4);
  unsigned int* wpk1 = (unsigned int*)take((size_t)9 * 7 * 4  * 256 * 4);
  unsigned int* wpk2 = (unsigned int*)take((size_t)9 * 2 * 4  * 256 * 4);
  unsigned int* wpk3 = (unsigned int*)take((size_t)9 * 2 * 4  * 256 * 4);
  unsigned int* wpk4 = (unsigned int*)take((size_t)9 * 2 * 16 * 256 * 4);

  // Zero halo-padded intermediates (ws is poisoned; halos must read as 0).
  const long zdw = (long)(actElems * 3 / 2);           // act1..act3 contiguous dwords
  zero_f16_kernel<<<(int)((zdw + 255) / 256), 256, 0, stream>>>((unsigned int*)act1, zdw);

  const long catN = (long)BB * HP * WP * CP0;
  pack_cat_kernel<<<(int)((catN + 255) / 256), 256, 0, stream>>>(x0, x1, x2, f1, f2, act0);
  pack_w_kernel<<<(9 * 7 * 4  * 256 + 255) / 256, 256, 0, stream>>>(w1, wpk1, CIN0, 7, 64, 4);
  pack_w_kernel<<<(9 * 2 * 4  * 256 + 255) / 256, 256, 0, stream>>>(w2, wpk2, 64, 2, 64, 4);
  pack_w_kernel<<<(9 * 2 * 4  * 256 + 255) / 256, 256, 0, stream>>>(w3, wpk3, 64, 2, 64, 4);
  pack_w_kernel<<<(9 * 2 * 16 * 256 + 255) / 256, 256, 0, stream>>>(w4, wpk4, 64, 2, OC4, 16);

  dim3 blk(128);
  conv_gemm_kernel<7, true,  false><<<dim3(NPIX / 128, 1), blk, 0, stream>>>(act0, wpk1, b1, act1, nullptr, 4, 64, 64);
  conv_gemm_kernel<2, true,  false><<<dim3(NPIX / 128, 1), blk, 0, stream>>>(act1, wpk2, b2, act2, nullptr, 4, 64, 64);
  conv_gemm_kernel<2, true,  false><<<dim3(NPIX / 128, 1), blk, 0, stream>>>(act2, wpk3, b3, act3, nullptr, 4, 64, 64);
  conv_gemm_kernel<2, false, true ><<<dim3(NPIX / 128, 4), blk, 0, stream>>>(act3, wpk4, b4, nullptr, o4, 16, OC4, NP4);

  deform_kernel<<<NPIX / 4, 256, 0, stream>>>(x0, o4, f1, f2, wd, bd, out);
}
